// GraphEmbeddings_70222715290046
// MI455X (gfx1250) — compile-verified, matrix operand order checked by reading, not run on hardware
//
#include <hip/hip_runtime.h>
#include <hip/hip_bf16.h>

typedef __attribute__((ext_vector_type(16))) _Float16 v16h;
typedef __attribute__((ext_vector_type(8)))  float    v8f;

#define Hh 2
#define Dd 128
#define NEG_ATT 0.2f
#define NEG_OUT 0.01f
#define BN_EPS 1e-5f

// float atomic max via integer atomics (lowers to native global_atomic_max_i32 /
// global_atomic_min_u32, no CAS loop). Valid given init value -FLT_MAX.
__device__ __forceinline__ void atomicMaxF(float* addr, float v) {
    if (v >= 0.0f) atomicMax((int*)addr, __float_as_int(v));
    else           atomicMin((unsigned int*)addr, __float_as_uint(v));
}

// ---------------- K0: init accumulators ----------------
__global__ void k_init(float* amax, float* denom, float* S1, float* S0,
                       float* mom, float* out, int NH, int outN) {
    int i = blockIdx.x * blockDim.x + threadIdx.x;
    if (i < NH) { amax[i] = -3.402823466e38f; denom[i] = 0.f; S1[i] = 0.f; S0[i] = 0.f; }
    if (i < 16)  mom[i] = 0.f;
    if (i < outN) out[i] = 0.f;
}

// ---------------- K1: per-(edge,head) raw attention score + scatter-max ----------------
__global__ void k_edge_alpha(const float* __restrict__ x, const long long* __restrict__ ei,
                             const float* __restrict__ eattr,
                             const float* __restrict__ Wl, const float* __restrict__ bl,
                             const float* __restrict__ Wr, const float* __restrict__ br,
                             const float* __restrict__ We, const float* __restrict__ att,
                             float* __restrict__ ea, float* __restrict__ amax, long long E) {
    __shared__ float sWl[Hh*Dd], sWr[Hh*Dd], sWe[Hh*Dd], sB[Hh*Dd], sAtt[Hh*Dd];
    int t = threadIdx.x;           // blockDim.x == 256 == Hh*Dd
    sWl[t] = Wl[t]; sWr[t] = Wr[t]; sWe[t] = We[t];
    sB[t] = bl[t] + br[t]; sAtt[t] = att[t];
    __syncthreads();
    long long tid = (long long)blockIdx.x * blockDim.x + t;
    if (tid >= E * Hh) return;
    long long e = tid >> 1; int h = (int)(tid & 1);
    long long s = ei[e], d = ei[E + e];
    float xs = x[s], xd = x[d], ae = eattr[e];
    float acc = 0.f;
    int base = h * Dd;
    #pragma unroll 8
    for (int k = 0; k < Dd; ++k) {
        int kk = base + k;
        float v = fmaf(xs, sWl[kk], fmaf(xd, sWr[kk], fmaf(ae, sWe[kk], sB[kk])));
        v = v > 0.f ? v : NEG_ATT * v;
        acc = fmaf(v, sAtt[kk], acc);
    }
    ea[tid] = acc;
    atomicMaxF(&amax[d * Hh + h], acc);
}

// ---------------- K2: exp(alpha - max) + scatter-sum denominator ----------------
__global__ void k_edge_exp(const long long* __restrict__ ei, float* __restrict__ ea,
                           const float* __restrict__ amax, float* __restrict__ denom, long long E) {
    long long tid = (long long)blockIdx.x * blockDim.x + threadIdx.x;
    if (tid >= E * Hh) return;
    long long e = tid >> 1; int h = (int)(tid & 1);
    long long d = ei[E + e];
    float v = __expf(ea[tid] - amax[d * Hh + h]);
    ea[tid] = v;
    atomicAdd(&denom[d * Hh + h], v);
}

// ---------------- K3: normalized weight -> factored aggregation (S1, S0) ----------------
__global__ void k_edge_agg(const float* __restrict__ x, const long long* __restrict__ ei,
                           const float* __restrict__ ea, const float* __restrict__ denom,
                           float* __restrict__ S1, float* __restrict__ S0, long long E) {
    long long tid = (long long)blockIdx.x * blockDim.x + threadIdx.x;
    if (tid >= E * Hh) return;
    long long e = tid >> 1; int h = (int)(tid & 1);
    long long s = ei[e], d = ei[E + e];
    float w = ea[tid] / (denom[d * Hh + h] + 1e-16f);
    atomicAdd(&S1[d * Hh + h], w * x[s]);
    atomicAdd(&S0[d * Hh + h], w);
}

// ---------------- K4: 14-value moment reduction (BN stats via u = 4-vector) ----------------
__global__ void k_moments(const float* __restrict__ S1, const float* __restrict__ S0,
                          float* __restrict__ mom, int N) {
    int n = blockIdx.x * blockDim.x + threadIdx.x;
    float u[4] = {0.f, 0.f, 0.f, 0.f};
    if (n < N) { u[0] = S1[2*n]; u[1] = S0[2*n]; u[2] = S1[2*n+1]; u[3] = S0[2*n+1]; }
    float v[14];
    v[0] = u[0]; v[1] = u[1]; v[2] = u[2]; v[3] = u[3];
    int idx = 4;
    for (int i = 0; i < 4; ++i)
        for (int j = i; j < 4; ++j) v[idx++] = u[i] * u[j];
    // wave32 shuffle reduction
    for (int off = 16; off > 0; off >>= 1)
        #pragma unroll
        for (int i = 0; i < 14; ++i) v[i] += __shfl_down(v[i], off, 32);
    if ((threadIdx.x & 31) == 0)
        #pragma unroll
        for (int i = 0; i < 14; ++i) atomicAdd(&mom[i], v[i]);
}

// ---------------- K5: fold BN affine into channel matrix B'[4][128] and t[128] ----------------
__global__ void k_finalize(const float* __restrict__ mom,
                           const float* __restrict__ Wl, const float* __restrict__ bl,
                           const float* __restrict__ bias, const float* __restrict__ gamma,
                           const float* __restrict__ beta,
                           float* __restrict__ Bp, float* __restrict__ tv, float invN) {
    int d = threadIdx.x;  // 128 threads
    float ub[4];
    #pragma unroll
    for (int k = 0; k < 4; ++k) ub[k] = mom[k] * invN;
    float cov[4][4];
    int idx = 4;
    for (int i = 0; i < 4; ++i)
        for (int j = i; j < 4; ++j) {
            float c2 = mom[idx++] * invN - ub[i] * ub[j];
            cov[i][j] = c2; cov[j][i] = c2;
        }
    float c[4] = {0.5f * Wl[d], 0.5f * bl[d], 0.5f * Wl[Dd + d], 0.5f * bl[Dd + d]};
    float mean = bias[d];
    #pragma unroll
    for (int k = 0; k < 4; ++k) mean = fmaf(c[k], ub[k], mean);
    float var = 0.f;
    for (int i = 0; i < 4; ++i)
        for (int j = 0; j < 4; ++j) var = fmaf(c[i] * cov[i][j], c[j], var);
    float scale = gamma[d] * rsqrtf(var + BN_EPS);
    tv[d] = fmaf(scale, bias[d] - mean, beta[d]);
    #pragma unroll
    for (int k = 0; k < 4; ++k) Bp[k * Dd + d] = scale * c[k];
}

// ---------------- K6: WMMA [16-node x 4] x [4 x 16-chan] + t, LeakyReLU, pool ----------------
// K padded 4->32 with zeros. A layout (16-bit, 16x32): lanes 0-15 = row M=lane,
// VGPR0 holds K=0,1 and VGPR1 holds K=2,3 -> v16h elements [0..3]; lanes 16-31 carry
// K=8..15/24..31 -> all zero here. B (32x16): lanes 0-15 = col N=lane, K=0..3 in
// elements [0..3]; lanes 16-31 (K=8..15/24..31) zero. C/D: lane -> N=lane&15, VGPR r -> M=r+(lane>=16?8:0).
__global__ void k_pool(const float* __restrict__ S1, const float* __restrict__ S0,
                       const float* __restrict__ Bp, const float* __restrict__ tv,
                       const long long* __restrict__ batch, float* __restrict__ out, int N) {
    int tile = blockIdx.x;            // 16 nodes per tile
    int wv   = threadIdx.x >> 5;      // 8 waves -> 8 column tiles of 16 channels
    int lane = threadIdx.x & 31;

    v16h a = {};
    if (lane < 16) {
        int node = tile * 16 + lane;
        if (node < N) {
            a[0] = (_Float16)S1[2 * node];
            a[1] = (_Float16)S0[2 * node];
            a[2] = (_Float16)S1[2 * node + 1];
            a[3] = (_Float16)S0[2 * node + 1];
        }
    }
    v16h b = {};
    if (lane < 16) {
        int col = wv * 16 + lane;
        b[0] = (_Float16)Bp[0 * Dd + col];
        b[1] = (_Float16)Bp[1 * Dd + col];
        b[2] = (_Float16)Bp[2 * Dd + col];
        b[3] = (_Float16)Bp[3 * Dd + col];
    }
    int colg = wv * 16 + (lane & 15);
    float tvc = tv[colg];
    v8f c = {tvc, tvc, tvc, tvc, tvc, tvc, tvc, tvc};

    v8f dres = __builtin_amdgcn_wmma_f32_16x16x32_f16(
        /*neg_a=*/false, a, /*neg_b=*/false, b,
        /*c_mod=*/(short)0, c, /*reuse_a=*/false, /*reuse_b=*/false);

    int rowbase = (lane & 16) ? 8 : 0;
    #pragma unroll
    for (int r = 0; r < 8; ++r) {
        int node = tile * 16 + rowbase + r;
        if (node < N) {
            float y = dres[r];
            y = y > 0.f ? y : NEG_OUT * y;
            int g = (int)batch[node];
            atomicAdd(&out[g * Dd + colg], y);
        }
    }
}

extern "C" void kernel_launch(void* const* d_in, const int* in_sizes, int n_in,
                              void* d_out, int out_size, void* d_ws, size_t ws_size,
                              hipStream_t stream) {
    const float*     x     = (const float*)d_in[0];
    const long long* ei    = (const long long*)d_in[1];
    const float*     eattr = (const float*)d_in[2];
    const long long* batch = (const long long*)d_in[3];
    const float*     Wl    = (const float*)d_in[4];
    const float*     bl    = (const float*)d_in[5];
    const float*     Wr    = (const float*)d_in[6];
    const float*     br    = (const float*)d_in[7];
    const float*     We    = (const float*)d_in[8];
    const float*     att   = (const float*)d_in[9];
    const float*     bias  = (const float*)d_in[10];
    const float*     gamma = (const float*)d_in[11];
    const float*     beta  = (const float*)d_in[12];
    float*           out   = (float*)d_out;

    const int       N  = in_sizes[0];
    const long long E  = in_sizes[2];
    const int       NH = N * Hh;

    // workspace layout (floats)
    float* ws    = (float*)d_ws;
    float* ea    = ws;                         // E*H
    float* amax  = ea + (size_t)E * Hh;        // N*H
    float* denom = amax + NH;                  // N*H
    float* S1    = denom + NH;                 // N*H
    float* S0    = S1 + NH;                    // N*H
    float* mom   = S0 + NH;                    // 16
    float* Bp    = mom + 16;                   // 4*128
    float* tvv   = Bp + 4 * Dd;                // 128

    k_init<<<(NH + 255) / 256, 256, 0, stream>>>(amax, denom, S1, S0, mom, out, NH, out_size);

    long long nEdgeThreads = E * Hh;
    int edgeBlocks = (int)((nEdgeThreads + 255) / 256);
    k_edge_alpha<<<edgeBlocks, 256, 0, stream>>>(x, ei, eattr, Wl, bl, Wr, br, We, att, ea, amax, E);
    k_edge_exp  <<<edgeBlocks, 256, 0, stream>>>(ei, ea, amax, denom, E);
    k_edge_agg  <<<edgeBlocks, 256, 0, stream>>>(x, ei, ea, denom, S1, S0, E);

    k_moments<<<(N + 255) / 256, 256, 0, stream>>>(S1, S0, mom, N);
    k_finalize<<<1, Dd, 0, stream>>>(mom, Wl, bl, bias, gamma, beta, Bp, tvv, 1.0f / (float)N);
    k_pool<<<(N + 15) / 16, 256, 0, stream>>>(S1, S0, Bp, tvv, batch, out, N);
}
